// EternalRecursion_80161269613291
// MI455X (gfx1250) — matrix operand order repaired and checked by reading, not compiled, not run on hardware
//
#include <hip/hip_runtime.h>
#include <hip/hip_bf16.h>
#include <math.h>

// ---------------------------------------------------------------------------
// GRU recurrence, B=512, D=500, 64 steps.  bf16 hi/lo split WMMA GEMMs.
// One fused kernel per step:
//   - done(t) recomputed from deterministic int64 fixed-point per-step sums
//   - epilogue latches h and writes the bf16 split state (ping/pong buffers)
//   - step 0 special-cased (h == 0 -> gh = b_hh, gi-only GEMM)
// ---------------------------------------------------------------------------

#define GRU_D   500
#define GRU_B   512
#define KP      512     // padded K (inner dim)
#define NP      512     // padded per-gate output columns
#define STEPS   64
#define NWAVES  512     // 16 M-groups (32 rows) x 32 N-tiles
#define FIXSCALE 1048576.f   // 2^20 fixed-point scale for deterministic mean

typedef __bf16 bf16_t;
typedef __attribute__((ext_vector_type(16))) __bf16 v16bf;
typedef __attribute__((ext_vector_type(8)))  __bf16 v8bf;
typedef __attribute__((ext_vector_type(8)))  float  v8f;

// ---- fragment loaders -----------------------------------------------------
__device__ __forceinline__ v16bf ldfrag(const bf16_t* p0, const bf16_t* p1) {
    v8bf a = *reinterpret_cast<const v8bf*>(p0);
    v8bf b = *reinterpret_cast<const v8bf*>(p1);
    v16bf r;
#pragma unroll
    for (int i = 0; i < 8; ++i) { r[i] = a[i]; r[i + 8] = b[i]; }
    return r;
}

__device__ __forceinline__ v8f wmma_bf16(v16bf a, v16bf b, v8f c) {
    return __builtin_amdgcn_wmma_f32_16x16x32_bf16(
        /*neg_a=*/false, a, /*neg_b=*/false, b,
        /*c_mod=*/(short)0, c, /*reuse_a=*/false, /*reuse_b=*/false);
}

// hi/lo split product: acc += A*B with A = a_hi+a_lo, B = b_hi+b_lo (drop lo*lo)
__device__ __forceinline__ v8f split_mma(v16bf a_hi, v16bf a_lo,
                                         v16bf b_hi, v16bf b_lo, v8f acc) {
    acc = wmma_bf16(a_lo, b_hi, acc);
    acc = wmma_bf16(a_hi, b_lo, acc);
    acc = wmma_bf16(a_hi, b_hi, acc);
    return acc;
}

__device__ __forceinline__ float sigm(float x) { return 1.f / (1.f + expf(-x)); }

// done(t) = OR over previous steps' means (deterministic fixed-point sums)
__device__ __forceinline__ bool done_before(int t, const unsigned long long* sums, float bcv) {
    bool done = false;
    const float inv = 1.f / (FIXSCALE * (float)(GRU_B * GRU_D));
    for (int q = 0; q < t; ++q) {
        float mean = (float)(long long)sums[q] * inv;
        done = done || (mean > bcv);
    }
    return done;
}

// ---- setup: h = 0, ping = split(state), pong padding = 0, sums = 0 --------
__global__ void gru_setup_kernel(const float* __restrict__ state,
                                 float* __restrict__ h,
                                 bf16_t* __restrict__ Aping_hi, bf16_t* __restrict__ Aping_lo,
                                 bf16_t* __restrict__ Apong_hi, bf16_t* __restrict__ Apong_lo,
                                 unsigned long long* __restrict__ sums) {
    int i = blockIdx.x * blockDim.x + threadIdx.x;       // over B*KP
    if (i >= GRU_B * KP) return;
    int r = i / KP, j = i % KP;
    if (j < GRU_D) {
        float v = state[r * GRU_D + j];
        h[r * GRU_D + j] = 0.f;
        bf16_t hi = (bf16_t)v;
        Aping_hi[i] = hi;
        Aping_lo[i] = (bf16_t)(v - (float)hi);
    } else {
        Aping_hi[i] = (bf16_t)0.f;
        Aping_lo[i] = (bf16_t)0.f;
        Apong_hi[i] = (bf16_t)0.f;
        Apong_lo[i] = (bf16_t)0.f;
    }
    if (i < STEPS) sums[i] = 0ull;
}

// ---- weights -> padded [3][NP][KP] bf16 hi/lo (per-gate zero padding) -----
__global__ void gru_split_weights_kernel(const float* __restrict__ W_ih,
                                         const float* __restrict__ W_hh,
                                         bf16_t* __restrict__ Wih_hi, bf16_t* __restrict__ Wih_lo,
                                         bf16_t* __restrict__ Whh_hi, bf16_t* __restrict__ Whh_lo) {
    long long idx = (long long)blockIdx.x * blockDim.x + threadIdx.x;
    const long long per_mat = 3LL * NP * KP;
    if (idx >= 2 * per_mat) return;
    int mat = (int)(idx / per_mat);
    long long r = idx % per_mat;
    int g = (int)(r / (NP * KP));
    int j = (int)((r / KP) % NP);
    int k = (int)(r % KP);
    const float* W = mat ? W_hh : W_ih;
    float v = (j < GRU_D && k < GRU_D) ? W[(long long)(g * GRU_D + j) * GRU_D + k] : 0.f;
    bf16_t hi = (bf16_t)v;
    bf16_t lo = (bf16_t)(v - (float)hi);
    bf16_t* dh = mat ? Whh_hi : Wih_hi;
    bf16_t* dl = mat ? Whh_lo : Wih_lo;
    dh[r] = hi;
    dl[r] = lo;
}

// ---- step 0: h == 0 -> gh = b_hh; gi via WMMA; done == false --------------
__global__ __launch_bounds__(256)
void gru_step0_kernel(const bf16_t* __restrict__ A_hi, const bf16_t* __restrict__ A_lo,
                      const bf16_t* __restrict__ Wih_hi, const bf16_t* __restrict__ Wih_lo,
                      const float* __restrict__ b_ih, const float* __restrict__ b_hh,
                      float* __restrict__ h,
                      bf16_t* __restrict__ Apong_hi, bf16_t* __restrict__ Apong_lo,
                      unsigned long long* __restrict__ sums) {
    const int wave = (blockIdx.x * blockDim.x + threadIdx.x) >> 5;  // 0..511
    const int lane = threadIdx.x & 31;
    const int mg = wave >> 5, tn = wave & 31;
    const int m0 = mg * 32, j0 = tn * 16;

    v8f accI[3][2] = {{{}, {}}, {{}, {}}, {{}, {}}};

    const int mrow = lane & 15;
    const int hsel = lane >> 4;
    const long long aoff0 = (long long)(m0 + mrow) * KP + 8 * hsel;
    const long long aoff1 = aoff0 + 16LL * KP;
    const long long bcol  = (long long)(j0 + mrow) * KP + 8 * hsel;

    for (int kt = 0; kt < KP; kt += 32) {
        v16bf a0_hi = ldfrag(A_hi + aoff0 + kt, A_hi + aoff0 + kt + 16);
        v16bf a0_lo = ldfrag(A_lo + aoff0 + kt, A_lo + aoff0 + kt + 16);
        v16bf a1_hi = ldfrag(A_hi + aoff1 + kt, A_hi + aoff1 + kt + 16);
        v16bf a1_lo = ldfrag(A_lo + aoff1 + kt, A_lo + aoff1 + kt + 16);
#pragma unroll
        for (int g = 0; g < 3; ++g) {
            const long long bo = (long long)g * NP * KP + bcol + kt;
            v16bf bih_hi = ldfrag(Wih_hi + bo, Wih_hi + bo + 16);
            v16bf bih_lo = ldfrag(Wih_lo + bo, Wih_lo + bo + 16);
            accI[g][0] = split_mma(a0_hi, a0_lo, bih_hi, bih_lo, accI[g][0]);
            accI[g][1] = split_mma(a1_hi, a1_lo, bih_hi, bih_lo, accI[g][1]);
        }
    }

    const int n = lane & 15;
    const int j = j0 + n;
    const bool valid = (j < GRU_D);
    float br = 0.f, bz = 0.f, bni = 0.f, bnh = 0.f;
    if (valid) {
        br  = b_ih[j]            + b_hh[j];
        bz  = b_ih[GRU_D + j]    + b_hh[GRU_D + j];
        bni = b_ih[2 * GRU_D + j];
        bnh = b_hh[2 * GRU_D + j];
    }
    const int rsel = lane >> 4;
    float lsum = 0.f;
#pragma unroll
    for (int t = 0; t < 2; ++t) {
#pragma unroll
        for (int v = 0; v < 8; ++v) {
            int m = m0 + t * 16 + v + 8 * rsel;
            float r  = sigm(accI[0][t][v] + br);
            float z  = sigm(accI[1][t][v] + bz);
            float nn = tanhf(accI[2][t][v] + bni + r * bnh);
            float hv = (1.f - z) * nn;                      // h_prev == 0
            if (valid) {
                h[(long long)m * GRU_D + j] = hv;           // done==false at step 0
                bf16_t hi = (bf16_t)hv;
                Apong_hi[(long long)m * KP + j] = hi;
                Apong_lo[(long long)m * KP + j] = (bf16_t)(hv - (float)hi);
                lsum += hv;
            }
        }
    }
#pragma unroll
    for (int off = 16; off > 0; off >>= 1) lsum += __shfl_down(lsum, off, 32);
    if (lane == 0)
        atomicAdd(&sums[0], (unsigned long long)(long long)llrintf(lsum * FIXSCALE));
}

// ---- main fused step kernel (steps >= 1): s == h == cur -------------------
__global__ __launch_bounds__(256)
void gru_step_kernel(int step,
                     const bf16_t* __restrict__ A_hi, const bf16_t* __restrict__ A_lo,
                     const bf16_t* __restrict__ Wih_hi, const bf16_t* __restrict__ Wih_lo,
                     const bf16_t* __restrict__ Whh_hi, const bf16_t* __restrict__ Whh_lo,
                     const float* __restrict__ b_ih, const float* __restrict__ b_hh,
                     float* __restrict__ h,
                     bf16_t* __restrict__ Apong_hi, bf16_t* __restrict__ Apong_lo,
                     unsigned long long* __restrict__ sums,
                     const float* __restrict__ bc) {
    const int wave = (blockIdx.x * blockDim.x + threadIdx.x) >> 5;  // 0..511
    const int lane = threadIdx.x & 31;
    const int mg = wave >> 5, tn = wave & 31;
    const int m0 = mg * 32, j0 = tn * 16;

    const bool done = done_before(step, sums, bc[0]);

    v8f accI[3][2] = {{{}, {}}, {{}, {}}, {{}, {}}};   // cur @ W_ih^T
    v8f accH[3][2] = {{{}, {}}, {{}, {}}, {{}, {}}};   // cur @ W_hh^T

    const int mrow = lane & 15;
    const int hsel = lane >> 4;
    const long long aoff0 = (long long)(m0 + mrow) * KP + 8 * hsel;
    const long long aoff1 = aoff0 + 16LL * KP;
    const long long bcol  = (long long)(j0 + mrow) * KP + 8 * hsel;

    for (int kt = 0; kt < KP; kt += 32) {
        v16bf a0_hi = ldfrag(A_hi + aoff0 + kt, A_hi + aoff0 + kt + 16);
        v16bf a0_lo = ldfrag(A_lo + aoff0 + kt, A_lo + aoff0 + kt + 16);
        v16bf a1_hi = ldfrag(A_hi + aoff1 + kt, A_hi + aoff1 + kt + 16);
        v16bf a1_lo = ldfrag(A_lo + aoff1 + kt, A_lo + aoff1 + kt + 16);
#pragma unroll
        for (int g = 0; g < 3; ++g) {
            const long long bo = (long long)g * NP * KP + bcol + kt;
            v16bf bih_hi = ldfrag(Wih_hi + bo, Wih_hi + bo + 16);
            v16bf bih_lo = ldfrag(Wih_lo + bo, Wih_lo + bo + 16);
            accI[g][0] = split_mma(a0_hi, a0_lo, bih_hi, bih_lo, accI[g][0]);
            accI[g][1] = split_mma(a1_hi, a1_lo, bih_hi, bih_lo, accI[g][1]);
            v16bf bhh_hi = ldfrag(Whh_hi + bo, Whh_hi + bo + 16);
            v16bf bhh_lo = ldfrag(Whh_lo + bo, Whh_lo + bo + 16);
            accH[g][0] = split_mma(a0_hi, a0_lo, bhh_hi, bhh_lo, accH[g][0]);
            accH[g][1] = split_mma(a1_hi, a1_lo, bhh_hi, bhh_lo, accH[g][1]);
        }
    }

    const int n = lane & 15;
    const int j = j0 + n;
    const bool valid = (j < GRU_D);
    float br = 0.f, bz = 0.f, bni = 0.f, bnh = 0.f;
    if (valid) {
        br  = b_ih[j]            + b_hh[j];
        bz  = b_ih[GRU_D + j]    + b_hh[GRU_D + j];
        bni = b_ih[2 * GRU_D + j];
        bnh = b_hh[2 * GRU_D + j];
    }
    const int rsel = lane >> 4;
    float lsum = 0.f;
#pragma unroll
    for (int t = 0; t < 2; ++t) {
#pragma unroll
        for (int v = 0; v < 8; ++v) {
            int m = m0 + t * 16 + v + 8 * rsel;
            float r  = sigm(accI[0][t][v] + accH[0][t][v] + br);
            float z  = sigm(accI[1][t][v] + accH[1][t][v] + bz);
            float nn = tanhf(accI[2][t][v] + bni + r * (accH[2][t][v] + bnh));
            if (valid) {
                float hp = h[(long long)m * GRU_D + j];     // this thread owns (m,j)
                float hv = (1.f - z) * nn + z * hp;
                lsum += hv;                                  // mean always uses h_new
                float keep = done ? hp : hv;                 // latch with OLD done
                h[(long long)m * GRU_D + j] = keep;
                bf16_t hi = (bf16_t)keep;
                Apong_hi[(long long)m * KP + j] = hi;
                Apong_lo[(long long)m * KP + j] = (bf16_t)(keep - (float)hi);
            }
        }
    }
#pragma unroll
    for (int off = 16; off > 0; off >>= 1) lsum += __shfl_down(lsum, off, 32);
    if (lane == 0)
        atomicAdd(&sums[step], (unsigned long long)(long long)llrintf(lsum * FIXSCALE));
}

__global__ void gru_copy_out_kernel(const float* __restrict__ h, float* __restrict__ out) {
    int i = blockIdx.x * blockDim.x + threadIdx.x;
    if (i < GRU_B * GRU_D) out[i] = h[i];
}

// ---------------------------------------------------------------------------
extern "C" void kernel_launch(void* const* d_in, const int* in_sizes, int n_in,
                              void* d_out, int out_size, void* d_ws, size_t ws_size,
                              hipStream_t stream) {
    const float* state = (const float*)d_in[0];
    const float* W_ih  = (const float*)d_in[1];
    const float* W_hh  = (const float*)d_in[2];
    const float* b_ih  = (const float*)d_in[3];
    const float* b_hh  = (const float*)d_in[4];
    const float* bc    = (const float*)d_in[5];
    (void)in_sizes; (void)n_in; (void)out_size;

    // workspace carve-up (256B aligned sections)
    char* ws = (char*)d_ws;
    size_t off = 0;
    auto carve = [&](size_t bytes) -> char* {
        char* p = ws + off;
        off += (bytes + 255) & ~(size_t)255;
        return p;
    };
    float*  h_buf   = (float*)carve(sizeof(float) * GRU_B * GRU_D);
    unsigned long long* sums = (unsigned long long*)carve(sizeof(unsigned long long) * STEPS);
    bf16_t* A_hi[2], *A_lo[2];
    A_hi[0] = (bf16_t*)carve(sizeof(bf16_t) * GRU_B * KP);
    A_lo[0] = (bf16_t*)carve(sizeof(bf16_t) * GRU_B * KP);
    A_hi[1] = (bf16_t*)carve(sizeof(bf16_t) * GRU_B * KP);
    A_lo[1] = (bf16_t*)carve(sizeof(bf16_t) * GRU_B * KP);
    bf16_t* Wih_hi = (bf16_t*)carve(sizeof(bf16_t) * 3 * NP * KP);
    bf16_t* Wih_lo = (bf16_t*)carve(sizeof(bf16_t) * 3 * NP * KP);
    bf16_t* Whh_hi = (bf16_t*)carve(sizeof(bf16_t) * 3 * NP * KP);
    bf16_t* Whh_lo = (bf16_t*)carve(sizeof(bf16_t) * 3 * NP * KP);
    if (off > ws_size) return;  // workspace too small; nothing safe to do

    const int NBD = (GRU_B * GRU_D + 255) / 256;           // 1000 blocks
    const int NBK = (GRU_B * KP + 255) / 256;              // 1024 blocks
    const int NWT = (int)((2LL * 3 * NP * KP + 255) / 256);
    const int NGB = NWAVES / 8;                            // 64 blocks x 8 waves

    gru_setup_kernel<<<NBK, 256, 0, stream>>>(state, h_buf, A_hi[0], A_lo[0],
                                              A_hi[1], A_lo[1], sums);
    gru_split_weights_kernel<<<NWT, 256, 0, stream>>>(W_ih, W_hh, Wih_hi, Wih_lo, Whh_hi, Whh_lo);

    // step 0: h == 0; reads ping(0), writes pong(1)
    gru_step0_kernel<<<NGB, 256, 0, stream>>>(A_hi[0], A_lo[0], Wih_hi, Wih_lo,
                                              b_ih, b_hh, h_buf, A_hi[1], A_lo[1], sums);

    // steps 1..63: fused GEMM + gates + latch + split + deterministic mean
    for (int step = 1; step < STEPS; ++step) {
        int p = step & 1;       // read buffer written by previous step
        gru_step_kernel<<<NGB, 256, 0, stream>>>(step,
                                                 A_hi[p], A_lo[p],
                                                 Wih_hi, Wih_lo, Whh_hi, Whh_lo,
                                                 b_ih, b_hh, h_buf,
                                                 A_hi[p ^ 1], A_lo[p ^ 1],
                                                 sums, bc);
    }
    gru_copy_out_kernel<<<NBD, 256, 0, stream>>>(h_buf, (float*)d_out);
}
